// GeometricBilinear_7567732375905
// MI455X (gfx1250) — compile-verified, matrix operand order checked by reading, not run on hardware
//
#include <hip/hip_runtime.h>

typedef __bf16 bf16_t;
typedef __attribute__((ext_vector_type(16))) __bf16 v16bf;
typedef __attribute__((ext_vector_type(8)))  float  v8f;

// ---------- helpers ----------
__device__ inline bf16_t f2bf(float f) { return (bf16_t)f; }   // RNE via hw cvt
__device__ inline float bf2f(bf16_t h) {
    unsigned u = ((unsigned)__builtin_bit_cast(unsigned short, h)) << 16;
    return __builtin_bit_cast(float, u);
}
// canonical blade-merge reordering sign parity (matches reference _sign)
__device__ __host__ constexpr int mergeSignBit(int a, int b) {
    int s = 0; a >>= 1;
    while (a) { s ^= (__builtin_popcount(a & b) & 1); a >>= 1; }
    return s;
}

// A/B fragment K-index pattern for 16-bit WMMA operands (ISA 7.12.2):
// element e of lane L covers K = base16(g) + 8*(L>=16) + (g&3)*2 + (e&1), g=e>>1
__device__ __host__ inline int kidx(int lane, int e) {
    int g = e >> 1;
    return ((g & 4) ? 16 : 0) + ((lane & 16) ? 8 : 0) + (g & 3) * 2 + (e & 1);
}

// ---------- weight pre-pack: f32 [O,I,9] -> bf16 WMMA A-fragments ----------
// a1: w_bilinear, layout [bsl(9)][ot(16)][ks(2)][lane(32)][e(16)]
// a2: w_next,     layout [bsl(9)][ot(4)][ks(4)][lane(32)][e(16)]
#define N_A1 (9 * 16 * 2 * 32 * 16)
#define N_A2 (9 * 4 * 4 * 32 * 16)

__global__ void pack_w(const float* __restrict__ wb, const float* __restrict__ wn,
                       bf16_t* __restrict__ a1, bf16_t* __restrict__ a2) {
    int idx = blockIdx.x * blockDim.x + threadIdx.x;
    if (idx < N_A1) {
        int e = idx & 15, lane = (idx >> 4) & 31, ks = (idx >> 9) & 1;
        int ot = (idx >> 10) & 15, bsl = idx >> 14;
        int k = ks * 32 + kidx(lane, e);
        int o = ot * 16 + (lane & 15);
        a1[idx] = f2bf(wb[(o * 64 + k) * 9 + bsl]);
    } else if (idx < N_A1 + N_A2) {
        int j = idx - N_A1;
        int e = j & 15, lane = (j >> 4) & 31, ks = (j >> 9) & 3;
        int ot = (j >> 11) & 3, bsl = j >> 13;
        int k = ks * 32 + kidx(lane, e);
        int o = ot * 16 + (lane & 15);
        a2[j] = f2bf(wn[(o * 128 + k) * 9 + bsl]);
    }
}

// ---------- LDS layout (single 160 KB arena, Of overlays dead Hh) ----------
#define SM_XTF   0                       // 16*2*32*16 bf16  = 32 KB
#define SM_HH    32768                   // 128*16*16 bf16   = 64 KB
#define SM_OF    32768                   // 16*64*16  f32    = 64 KB (reuses Hh)
#define SM_VF    98304                   // 16*4*32*16 bf16  = 64 KB
#define SM_REFP  163840                  // 16 f32
#define SM_TOTAL 163904

// ---------- fused main kernel: one 16-token tile per workgroup ----------
__global__ __launch_bounds__(256) void gatr_fused(
    const float* __restrict__ x, const float* __restrict__ ref,
    const float* __restrict__ bb, const float* __restrict__ bn,
    const bf16_t* __restrict__ a1, const bf16_t* __restrict__ a2,
    float* __restrict__ out)
{
    __shared__ __attribute__((aligned(32))) unsigned char smem[SM_TOTAL];
    bf16_t* XTf  = reinterpret_cast<bf16_t*>(smem + SM_XTF);
    bf16_t* Hh   = reinterpret_cast<bf16_t*>(smem + SM_HH);
    float*  Of   = reinterpret_cast<float*>(smem + SM_OF);
    bf16_t* VF   = reinterpret_cast<bf16_t*>(smem + SM_VF);
    float*  refp = reinterpret_cast<float*>(smem + SM_REFP);

    const int tid  = threadIdx.x;
    // wave-uniform scalar: forces SGPR control flow (full-EXEC WMMA, s-branches)
    const int wave = __builtin_amdgcn_readfirstlane(tid >> 5);
    const int lane = tid & 31;
    const long t0  = (long)blockIdx.x * 16;

    // ---- stage A: load x tile -> bf16 B-fragments in LDS ----
    if (tid < 16) refp[tid] = ref[(t0 + tid) * 16 + 15];
    // warm L2/WGP$ with the tail-GEMM weight fragments (global_prefetch_b8)
    __builtin_prefetch(a2 + (size_t)tid * 288, 0, 1);
    __builtin_prefetch(a2 + (size_t)tid * 288 + 144, 0, 1);
    for (int rep = 0; rep < 4; ++rep) {
        int job = tid + rep * 256;          // 1024 (token, in-ch) pairs
        int tt = job >> 6, i = job & 63;
        const float4* xp = reinterpret_cast<const float4*>(x + ((t0 + tt) * 64 + i) * 16);
        float4 q[4];
        q[0] = xp[0]; q[1] = xp[1]; q[2] = xp[2]; q[3] = xp[3];
        const float* vf = reinterpret_cast<const float*>(q);
        int ks = i >> 5, r = i & 31;
        int laneS = tt + ((r & 8) ? 16 : 0);
        int e = 2 * (((r & 16) ? 4 : 0) + ((r & 7) >> 1)) + (r & 1);
#pragma unroll
        for (int k = 0; k < 16; ++k)
            XTf[(((k * 2 + ks) * 32 + laneS) * 16) + e] = f2bf(vf[k]);
    }
    __syncthreads();

    // ---- stage B: first equi_linear, one 128-channel half -> Hh ----
    auto stageB = [&](int half) {
        for (int jj = 0; jj < 16; ++jj) {
            int job = jj * 8 + wave;         // scalar (wave came from readfirstlane)
            int otl = job >> 4;              // local out-tile 0..7
            int j   = job & 15;              // blade
            int ot  = half * 8 + otl;        // global out-tile 0..15
            int g   = __builtin_popcount(j);
            v8f acc = {};
#pragma unroll
            for (int ks = 0; ks < 2; ++ks) {
                v16bf A = *reinterpret_cast<const v16bf*>(
                    &a1[(((g * 16 + ot) * 2 + ks) * 32 + lane) * 16]);
                v16bf B = *reinterpret_cast<const v16bf*>(
                    &XTf[((j * 2 + ks) * 32 + lane) * 16]);
                acc = __builtin_amdgcn_wmma_f32_16x16x32_bf16(
                    false, A, false, B, (short)0, acc, false, false);
            }
            if (j & 1) {                     // e0-composed slices feed odd blades
                int bsl = 4 + g;             // = 5 + grade(j^1)
#pragma unroll
                for (int ks = 0; ks < 2; ++ks) {
                    v16bf A = *reinterpret_cast<const v16bf*>(
                        &a1[(((bsl * 16 + ot) * 2 + ks) * 32 + lane) * 16]);
                    v16bf B = *reinterpret_cast<const v16bf*>(
                        &XTf[(((j ^ 1) * 2 + ks) * 32 + lane) * 16]);
                    acc = __builtin_amdgcn_wmma_f32_16x16x32_bf16(
                        false, A, false, B, (short)0, acc, false, false);
                }
            }
            int t = lane & 15;
            int rofs = (lane & 16) ? 8 : 0;  // D layout: VGPR v -> M=v / v+8
            if (j == 0) {                    // bias on scalar blade, 8 contiguous f32
                const float4* bp = reinterpret_cast<const float4*>(
                    &bb[half * 128 + otl * 16 + rofs]);
                float4 b0 = bp[0], b1 = bp[1];
                acc[0] += b0.x; acc[1] += b0.y; acc[2] += b0.z; acc[3] += b0.w;
                acc[4] += b1.x; acc[5] += b1.y; acc[6] += b1.z; acc[7] += b1.w;
            }
#pragma unroll
            for (int v = 0; v < 8; ++v) {
                int cl = otl * 16 + v + rofs;        // local channel 0..127
                Hh[(cl * 16 + t) * 16 + j] = f2bf(acc[v]);
            }
        }
    };

    // scatter a 16-blade multivector into VF B-fragment order at K=cv, col=t
    auto writeVF = [&](int cv, int t, const float* acc) {
        int ks = cv >> 5, rr = cv & 31;
        int laneS = t + ((rr & 8) ? 16 : 0);
        int e = 2 * (((rr & 16) ? 4 : 0) + ((rr & 7) >> 1)) + (rr & 1);
#pragma unroll
        for (int k = 0; k < 16; ++k)
            VF[(((k * 4 + ks) * 32 + laneS) * 16) + e] = f2bf(acc[k]);
    };
    // fetch both operand multivectors of one (channel, token) job (vectorized ds)
    auto loadLR = [&](int c, int t, float* l, float* r) {
        v16bf lv = *reinterpret_cast<const v16bf*>(&Hh[(c * 16 + t) * 16]);
        v16bf rv = *reinterpret_cast<const v16bf*>(&Hh[((c + 64) * 16 + t) * 16]);
#pragma unroll
        for (int k = 0; k < 16; ++k) { l[k] = bf2f(lv[k]); r[k] = bf2f(rv[k]); }
    };

    stageB(0);
    __syncthreads();

    // ---- stage C1: geometric product (PGA, e0^2 = 0), per (ch,token) ----
    for (int rep = 0; rep < 4; ++rep) {
        int job = tid + rep * 256;
        int t = job & 15, c = job >> 4;      // c = 0..63
        float l[16], r[16];
        loadLR(c, t, l, r);
        float acc[16] = {};
#pragma unroll
        for (int a = 0; a < 16; ++a)
#pragma unroll
            for (int b = 0; b < 16; ++b)
                if (!(a & b & 1)) {          // degenerate metric kills shared-e0
                    float s = mergeSignBit(a, b) ? -1.f : 1.f;   // folds to const
                    acc[a ^ b] += s * l[a] * r[b];
                }
        writeVF(c, t, acc);
    }
    __syncthreads();

    stageB(1);                               // Hh now holds join-half channels
    __syncthreads();

    // ---- stage C2: equivariant join (dual-wedge-undual), scaled by ref pss ----
    for (int rep = 0; rep < 4; ++rep) {
        int job = tid + rep * 256;
        int t = job & 15, c = job >> 4;
        float l[16], r[16];
        loadLR(c, t, l, r);
        float rp = refp[t];
        float acc[16] = {};
#pragma unroll
        for (int a = 0; a < 16; ++a)
#pragma unroll
            for (int b = 0; b < 16; ++b)
                if ((a | b) == 15) {         // duals disjoint <=> a|b = pss
                    int kk = a & b;
                    int sp = mergeSignBit(a, 15 ^ a) ^ mergeSignBit(b, 15 ^ b) ^
                             mergeSignBit(15 ^ a, 15 ^ b) ^
                             mergeSignBit(kk, 15 ^ kk);
                    acc[kk] += (sp ? -1.f : 1.f) * l[a] * r[b];
                }
#pragma unroll
        for (int k = 0; k < 16; ++k) acc[k] *= rp;
        writeVF(64 + c, t, acc);
    }
    __syncthreads();                         // Hh dead from here -> Of overlays it

    // ---- stage E: second equi_linear (K=128) -> LDS out tile ----
    for (int jj = 0; jj < 8; ++jj) {
        int job = jj * 8 + wave;             // 64 jobs: 4 out-tiles x 16 blades
        int ot = job >> 4;
        int j  = job & 15;
        int g  = __builtin_popcount(j);
        v8f acc = {};
#pragma unroll
        for (int ks = 0; ks < 4; ++ks) {
            v16bf A = *reinterpret_cast<const v16bf*>(
                &a2[(((g * 4 + ot) * 4 + ks) * 32 + lane) * 16]);
            v16bf B = *reinterpret_cast<const v16bf*>(
                &VF[((j * 4 + ks) * 32 + lane) * 16]);
            acc = __builtin_amdgcn_wmma_f32_16x16x32_bf16(
                false, A, false, B, (short)0, acc, false, false);
        }
        if (j & 1) {
            int bsl = 4 + g;
#pragma unroll
            for (int ks = 0; ks < 4; ++ks) {
                v16bf A = *reinterpret_cast<const v16bf*>(
                    &a2[(((bsl * 4 + ot) * 4 + ks) * 32 + lane) * 16]);
                v16bf B = *reinterpret_cast<const v16bf*>(
                    &VF[(((j ^ 1) * 4 + ks) * 32 + lane) * 16]);
                acc = __builtin_amdgcn_wmma_f32_16x16x32_bf16(
                    false, A, false, B, (short)0, acc, false, false);
            }
        }
        int t = lane & 15;
        int rofs = (lane & 16) ? 8 : 0;
        if (j == 0) {
            const float4* bp = reinterpret_cast<const float4*>(&bn[ot * 16 + rofs]);
            float4 b0 = bp[0], b1 = bp[1];
            acc[0] += b0.x; acc[1] += b0.y; acc[2] += b0.z; acc[3] += b0.w;
            acc[4] += b1.x; acc[5] += b1.y; acc[6] += b1.z; acc[7] += b1.w;
        }
#pragma unroll
        for (int v = 0; v < 8; ++v) {
            int o = ot * 16 + v + rofs;
            Of[(t * 64 + o) * 16 + j] = acc[v];   // [token][o][blade] f32 tile
        }
    }
    __syncthreads();

    // ---- stage F: coalesced tile writeback (lane-contiguous b128 stores) ----
    {
        float4* gout = reinterpret_cast<float4*>(out + t0 * 64 * 16);
        const float4* lo = reinterpret_cast<const float4*>(Of);
#pragma unroll
        for (int rep = 0; rep < 16; ++rep)
            gout[rep * 256 + tid] = lo[rep * 256 + tid];
    }
}

extern "C" void kernel_launch(void* const* d_in, const int* in_sizes, int n_in,
                              void* d_out, int out_size, void* d_ws, size_t ws_size,
                              hipStream_t stream) {
    const float* x   = (const float*)d_in[0];
    const float* ref = (const float*)d_in[1];
    const float* wb  = (const float*)d_in[2];
    const float* bb  = (const float*)d_in[3];
    const float* wn  = (const float*)d_in[4];
    const float* bn  = (const float*)d_in[5];
    float* out = (float*)d_out;

    bf16_t* a1 = (bf16_t*)d_ws;
    bf16_t* a2 = a1 + N_A1;                  // ~432 KB total scratch

    const int tokens = in_sizes[0] / (64 * 16);   // B*S = 16384

    int prepThreads = N_A1 + N_A2;
    pack_w<<<(prepThreads + 255) / 256, 256, 0, stream>>>(wb, wn, a1, a2);
    gatr_fused<<<tokens / 16, 256, 0, stream>>>(x, ref, bb, bn, a1, a2, out);
}
